// RelativeSpatialAttention_17575006175534
// MI455X (gfx1250) — matrix operand order repaired and checked by reading, not compiled, not run on hardware
//
#include <hip/hip_runtime.h>

// Problem dims (fixed per reference)
#define S_DIM 1024
#define T_DIM 1024
#define B_DIM 4
#define C_DIM 1024
#define H_DIM 16
#define CC_DIM 64
#define M_DIM (S_DIM * B_DIM)   // 4096 rows, row m = s*B + b

typedef __attribute__((ext_vector_type(16))) __bf16 v16bf;
typedef __attribute__((ext_vector_type(8)))  __bf16 v8bf;
typedef __attribute__((ext_vector_type(8)))  float  v8f;

// ---- WMMA wrapper: D(f32 16x16) = A(bf16 16x32) x B(bf16 32x16) + C ----
static __device__ __forceinline__ v8f wmma_bf16(v16bf a, v16bf b, v8f c) {
  return __builtin_amdgcn_wmma_f32_16x16x32_bf16(false, a, false, b,
                                                 (short)0, c, false, false);
}

// A-matrix 16x32 bf16, contiguous k order used here (matches B) (per lane, khalf = lane>>4):
//   elems [0..7]  = K = khalf*16     + 0..7
//   elems [8..15] = K = khalf*16 + 8 + 0..7
static __device__ __forceinline__ v16bf load_a_from_f32(const float* __restrict__ row,
                                                        int kbase, int khalf) {
  const float* p0 = row + kbase + khalf * 16;
  const float* p1 = row + kbase + khalf * 16 + 8;
  v16bf o;
#pragma unroll
  for (int i = 0; i < 8; i += 4) {
    float4 c = *(const float4*)(p0 + i);
    o[i] = (__bf16)c.x; o[i + 1] = (__bf16)c.y; o[i + 2] = (__bf16)c.z; o[i + 3] = (__bf16)c.w;
  }
#pragma unroll
  for (int i = 0; i < 8; i += 4) {
    float4 c = *(const float4*)(p1 + i);
    o[8 + i] = (__bf16)c.x; o[9 + i] = (__bf16)c.y; o[10 + i] = (__bf16)c.z; o[11 + i] = (__bf16)c.w;
  }
  return o;
}

// B-matrix 32x16 bf16, contiguous k order used here, matching A (per lane): lanes 0-15 K=0..15, lanes 16-31 K=16..31,
// i.e. 16 consecutive K values from one W/kr row -> elems [0..15]
static __device__ __forceinline__ v16bf load_b_from_f32(const float* __restrict__ p) {
  v16bf o;
#pragma unroll
  for (int i = 0; i < 16; i += 4) {
    float4 c = *(const float4*)(p + i);
    o[i] = (__bf16)c.x; o[i + 1] = (__bf16)c.y; o[i + 2] = (__bf16)c.z; o[i + 3] = (__bf16)c.w;
  }
  return o;
}

// Works for both global and LDS source pointers (provenance decides ds vs vmem)
static __device__ __forceinline__ v16bf load_a_bf16(const __bf16* row, int kbase, int khalf) {
  v8bf c0 = *(const v8bf*)(row + kbase + khalf * 16);
  v8bf c1 = *(const v8bf*)(row + kbase + khalf * 16 + 8);
  v16bf o;
#pragma unroll
  for (int i = 0; i < 8; i++) { o[i] = c0[i]; o[8 + i] = c1[i]; }
  return o;
}

static __device__ __forceinline__ v16bf load_b_bf16(const __bf16* __restrict__ p) {
  v8bf c0 = *(const v8bf*)(p);
  v8bf c1 = *(const v8bf*)(p + 8);
  v16bf o;
#pragma unroll
  for (int i = 0; i < 8; i++) { o[i] = c0[i]; o[8 + i] = c1[i]; }
  return o;
}

// --------------------------------------------------------------------------
// Kernel 1: uwke = Wke@u+bke, vwke = Wkr@v+bkr (in LDS), then
//           pu = Wke^T@uwke, pv = Wkr^T@vwke, const0 = uwke.bke + vwke.bkr
// --------------------------------------------------------------------------
__global__ __launch_bounds__(256) void prep_kernel(
    const float* __restrict__ Wke, const float* __restrict__ bke,
    const float* __restrict__ Wkr, const float* __restrict__ bkr,
    const float* __restrict__ u, const float* __restrict__ v,
    float* __restrict__ pu, float* __restrict__ pv, float* __restrict__ c0out) {
  __shared__ float su[C_DIM];
  __shared__ float sv[C_DIM];
  __shared__ float red[256];
  const int tid = threadIdx.x;

  for (int c = tid; c < C_DIM; c += 256) {
    const float* wke_row = Wke + (size_t)c * C_DIM;
    const float* wkr_row = Wkr + (size_t)c * C_DIM;
    float du = 0.f, dv = 0.f;
    for (int k = 0; k < C_DIM; k++) { du += wke_row[k] * u[k]; dv += wkr_row[k] * v[k]; }
    su[c] = du + bke[c];
    sv[c] = dv + bkr[c];
  }
  __syncthreads();

  for (int k = tid; k < C_DIM; k += 256) {
    float du = 0.f, dv = 0.f;
    for (int c = 0; c < C_DIM; c++) {
      du += su[c] * Wke[(size_t)c * C_DIM + k];
      dv += sv[c] * Wkr[(size_t)c * C_DIM + k];
    }
    pu[k] = du; pv[k] = dv;
  }

  float part = 0.f;
  for (int c = tid; c < C_DIM; c += 256) part += su[c] * bke[c] + sv[c] * bkr[c];
  red[tid] = part; __syncthreads();
  for (int s = 128; s > 0; s >>= 1) { if (tid < s) red[tid] += red[tid + s]; __syncthreads(); }
  if (tid == 0) c0out[0] = red[0];
}

// --------------------------------------------------------------------------
// Kernel 2: bias[m] = x[m,:].pu + r[m,:].pv + const0   (m = t*B + b)
// --------------------------------------------------------------------------
__global__ __launch_bounds__(256) void bias_kernel(
    const float* __restrict__ x, const float* __restrict__ r,
    const float* __restrict__ pu, const float* __restrict__ pv,
    const float* __restrict__ c0, float* __restrict__ bias) {
  __shared__ float red[256];
  const int m = blockIdx.x;
  const int tid = threadIdx.x;
  const float* xrow = x + (size_t)m * C_DIM;
  const float* rrow = r + (size_t)m * C_DIM;
  float part = 0.f;
  for (int c = tid; c < C_DIM; c += 256) part += xrow[c] * pu[c] + rrow[c] * pv[c];
  red[tid] = part; __syncthreads();
  for (int s = 128; s > 0; s >>= 1) { if (tid < s) red[tid] += red[tid + s]; __syncthreads(); }
  if (tid == 0) bias[m] = red[0] + c0[0];
}

// --------------------------------------------------------------------------
// Kernel 3: projection GEMM (bf16 WMMA, f32 accumulate, bf16 output)
//   out[m,n] = sum_k A1[m,k]*W1[n,k] + bias1[n]  (+ A2/W2/bias2 if HAS2)
// Block: 256 thr = 8 waves -> tile 32m x 256n (wave = 16m x 64n), K-step 32.
// Templated on HAS2 so the inner loop is branch-free.
// --------------------------------------------------------------------------
template <bool HAS2>
__global__ __launch_bounds__(256, 1) void proj_gemm(
    const float* __restrict__ A1, const float* __restrict__ W1, const float* __restrict__ bias1,
    const float* __restrict__ A2, const float* __restrict__ W2, const float* __restrict__ bias2,
    unsigned short* __restrict__ out_raw) {
  __bf16* out = (__bf16*)out_raw;
  const int lane  = threadIdx.x & 31;
  const int wave  = threadIdx.x >> 5;
  const int m0    = blockIdx.y * 32 + (wave >> 2) * 16;
  const int n0    = blockIdx.x * 256 + (wave & 3) * 64;
  const int lm    = lane & 15;
  const int khalf = lane >> 4;

  const float* a1row = A1 + (size_t)(m0 + lm) * C_DIM;
  const float* a2row = HAS2 ? (A2 + (size_t)(m0 + lm) * C_DIM) : nullptr;

  v8f acc[4] = {};
  for (int k0 = 0; k0 < C_DIM; k0 += 32) {
    if (k0 + 64 < C_DIM) {
      __builtin_prefetch(a1row + k0 + 64, 0, 1);
      if (HAS2) __builtin_prefetch(a2row + k0 + 64, 0, 1);
    }
    v16bf a1 = load_a_from_f32(a1row, k0, khalf);
    v16bf a2;
    if (HAS2) a2 = load_a_from_f32(a2row, k0, khalf);
#pragma unroll
    for (int nt = 0; nt < 4; nt++) {
      const int n = n0 + nt * 16 + lm;
      v16bf b1 = load_b_from_f32(W1 + (size_t)n * C_DIM + k0 + khalf * 16);
      acc[nt] = wmma_bf16(a1, b1, acc[nt]);
      if (HAS2) {
        v16bf b2 = load_b_from_f32(W2 + (size_t)n * C_DIM + k0 + khalf * 16);
        acc[nt] = wmma_bf16(a2, b2, acc[nt]);
      }
    }
  }

  // C/D layout: VGPR j -> M = m0 + khalf*8 + j, N = n0 + nt*16 + lm
#pragma unroll
  for (int nt = 0; nt < 4; nt++) {
    const int n = n0 + nt * 16 + lm;
    const float bsum = HAS2 ? (bias1[n] + bias2[n]) : bias1[n];
#pragma unroll
    for (int j = 0; j < 8; j++) {
      const int m = m0 + khalf * 8 + j;
      out[(size_t)m * C_DIM + n] = (__bf16)(acc[nt][j] + bsum);
    }
  }
}

// --------------------------------------------------------------------------
// Kernel 4: attention terms. For each (b,h):
//   out[s,t,b,h] = ewq[s,b,h,:].kr[t,b,h,:] + bias[t,b]
// Block: 512 thr = 16 waves (one per head h), tile 16s x 64t, K = 64 (2 steps).
// A tile (16x64 bf16, wave-private) is staged to LDS with
// GLOBAL_LOAD_ASYNC_TO_LDS_B128 (ASYNCcnt), pulled into registers, then the
// same 64KB LDS is reused (post-barrier) to reorganize the 16x64x16 output
// tile for dense float4 stores (h is the innermost output dim).
// --------------------------------------------------------------------------
#define LIDX(sl, tl, h) ((((sl)*64 + (tl)) << 4) + (((h) + (tl)) & 15))

__global__ __launch_bounds__(512, 1) void attn_kernel(
    const unsigned short* __restrict__ ewq_raw, const unsigned short* __restrict__ kr_raw,
    const float* __restrict__ bias, float* __restrict__ out) {
  const __bf16* ewq = (const __bf16*)ewq_raw;
  const __bf16* kr  = (const __bf16*)kr_raw;
  // 64 KB. Phase 1: first 32KB = A-stage (2KB per wave, at h*2048).
  //        Phase 2 (after barrier): [s_local][t_local][h] swizzled out-stage.
  __shared__ float lds[16 * 64 * 16];

  const int lane  = threadIdx.x & 31;
  const int h     = threadIdx.x >> 5;     // 0..15: head per wave
  const int b     = blockIdx.z;
  const int s0    = blockIdx.y * 16;
  const int t0    = blockIdx.x * 64;
  const int lm    = lane & 15;
  const int khalf = lane >> 4;

  // ---- Phase 1a: async-stage the A tile (rows s0..s0+15, cols h*64..h*64+63)
  {
    const int row  = lane >> 1;   // 0..15
    const int half = lane & 1;    // 0..1
    const __bf16* gsrc = ewq + ((size_t)(s0 + row) * B_DIM + b) * C_DIM + h * CC_DIM;
    const unsigned lbase =
        (unsigned)(unsigned long long)(const void*)&lds[0] + h * 2048 + row * 128;
#pragma unroll
    for (int it = 0; it < 4; it++) {
      const int chunk = half + it * 2;  // 8 x 16B chunks per 128B row
      const unsigned laddr = lbase + chunk * 16;
      const unsigned long long gaddr =
          (unsigned long long)(const void*)gsrc + (unsigned)(chunk * 16);
      asm volatile("global_load_async_to_lds_b128 %0, %1, off"
                   :: "v"(laddr), "v"(gaddr) : "memory");
    }
  }
#if __has_builtin(__builtin_amdgcn_s_wait_asynccnt)
  __builtin_amdgcn_s_wait_asynccnt(0);
#else
  asm volatile("s_wait_asynccnt 0x0" ::: "memory");
#endif

  // ---- Phase 1b: pull A (both K-steps) from LDS into registers (ds loads)
  const __bf16* larow = (const __bf16*)lds + h * 1024 + lm * 64;  // bf16 units
  v16bf a_k[2];
#pragma unroll
  for (int ks = 0; ks < 2; ks++) a_k[ks] = load_a_bf16(larow, ks * 32, khalf);

  __syncthreads();  // all waves done reading A -> LDS reusable for out-stage

  // ---- Phase 2: K-loop (B tiles straight from global; they sit in L2)
  v8f acc[4] = {};
#pragma unroll
  for (int ks = 0; ks < 2; ks++) {
    const int k0 = ks * 32;
#pragma unroll
    for (int nt = 0; nt < 4; nt++) {
      const __bf16* brow = kr + ((size_t)(t0 + nt * 16 + lm) * B_DIM + b) * C_DIM
                              + h * CC_DIM + k0 + khalf * 16;
      acc[nt] = wmma_bf16(a_k[ks], load_b_bf16(brow), acc[nt]);
    }
  }

  // ---- Phase 3: stage 16x64x16 f32 tile, element (sl, tl, h), swizzled
#pragma unroll
  for (int nt = 0; nt < 4; nt++) {
#pragma unroll
    for (int j = 0; j < 8; j++) {
      lds[LIDX(khalf * 8 + j, nt * 16 + lm, h)] = acc[nt][j];
    }
  }
  __syncthreads();

  // ---- Phase 4: coalesced stores + bias
  for (int idx = threadIdx.x; idx < 16 * 64 * 4; idx += 512) {
    const int hq = idx & 3;
    const int tl = (idx >> 2) & 63;
    const int sl = idx >> 8;
    const float bval = bias[(t0 + tl) * B_DIM + b];
    float4 vv;
    vv.x = lds[LIDX(sl, tl, hq * 4 + 0)] + bval;
    vv.y = lds[LIDX(sl, tl, hq * 4 + 1)] + bval;
    vv.z = lds[LIDX(sl, tl, hq * 4 + 2)] + bval;
    vv.w = lds[LIDX(sl, tl, hq * 4 + 3)] + bval;
    float* dst = out + ((((size_t)(s0 + sl)) * T_DIM + (t0 + tl)) * B_DIM + b) * H_DIM + hq * 4;
    *(float4*)dst = vv;
  }
}

// --------------------------------------------------------------------------
extern "C" void kernel_launch(void* const* d_in, const int* in_sizes, int n_in,
                              void* d_out, int out_size, void* d_ws, size_t ws_size,
                              hipStream_t stream) {
  (void)in_sizes; (void)n_in; (void)out_size; (void)ws_size;
  const float* x   = (const float*)d_in[0];
  const float* r   = (const float*)d_in[1];
  const float* Wq  = (const float*)d_in[2];
  const float* bq  = (const float*)d_in[3];
  const float* Wke = (const float*)d_in[4];
  const float* bke = (const float*)d_in[5];
  const float* Wkr = (const float*)d_in[6];
  const float* bkr = (const float*)d_in[7];
  const float* u   = (const float*)d_in[8];
  const float* v   = (const float*)d_in[9];
  float* out = (float*)d_out;

  char* ws = (char*)d_ws;
  float*          bias = (float*)(ws);                         // 4096 f32 (t*B+b)
  float*          pu   = (float*)(ws + 16 * 1024);             // 1024 f32
  float*          pv   = (float*)(ws + 20 * 1024);             // 1024 f32
  float*          c0   = (float*)(ws + 24 * 1024);             // 1 f32
  unsigned short* ewq  = (unsigned short*)(ws + 32 * 1024);    // 4096x1024 bf16 (8 MB)
  unsigned short* kr   = (unsigned short*)(ws + 32 * 1024 + (size_t)M_DIM * C_DIM * 2);

  prep_kernel<<<1, 256, 0, stream>>>(Wke, bke, Wkr, bkr, u, v, pu, pv, c0);
  bias_kernel<<<M_DIM, 256, 0, stream>>>(x, r, pu, pv, c0, bias);
  // ewq = x @ Wq^T + bq
  proj_gemm<false><<<dim3(C_DIM / 256, M_DIM / 32), 256, 0, stream>>>(
      x, Wq, bq, nullptr, nullptr, nullptr, ewq);
  // kr = (x @ Wke^T + bke) + (r @ Wkr^T + bkr)   [term_a+term_b fusion]
  proj_gemm<true><<<dim3(C_DIM / 256, M_DIM / 32), 256, 0, stream>>>(
      x, Wke, bke, r, Wkr, bkr, kr);
  // out[s,t,b,h] = ewq . kr + bias[t,b]
  attn_kernel<<<dim3(T_DIM / 64, S_DIM / 16, B_DIM), 512, 0, stream>>>(
      ewq, kr, bias, out);
}
